// TTTBaseModule_3573412790639
// MI455X (gfx1250) — matrix operand order repaired and checked by reading, not compiled, not run on hardware
//
#include <hip/hip_runtime.h>

typedef unsigned short u16;
typedef unsigned int   u32;

typedef __attribute__((ext_vector_type(8)))  float  v8f;
typedef __attribute__((ext_vector_type(2)))  float  v2f;
typedef __attribute__((ext_vector_type(16))) __bf16 v16bf;

constexpr int cB  = 4;
constexpr int cL  = 2048;
constexpr int cD  = 2048;
constexpr int cNH = 32;
constexpr int cHD = 64;
constexpr int cK  = 16;
constexpr int cNC = 128;

// ---------------------------------------------------------------- helpers

__device__ inline u16 f32_to_bf16(float f) {
  u32 u = __float_as_uint(f);
  u32 r = u + 0x7FFFu + ((u >> 16) & 1u);   // RNE
  return (u16)(r >> 16);
}

__device__ inline v8f wmma_bf16x32(v16bf a, v16bf b, v8f c) {
  return __builtin_amdgcn_wmma_f32_16x16x32_bf16(false, a, false, b, (short)0, c,
                                                 false, false);
}

__device__ inline float redsum16(float s) {
  s += __shfl_xor(s, 8, 32);
  s += __shfl_xor(s, 4, 32);
  s += __shfl_xor(s, 2, 32);
  s += __shfl_xor(s, 1, 32);
  return s;
}

// ------- 16x16x4 f32 WMMA (preferred) with bf16 zero-padded fallback -----
#if defined(__has_builtin)
#if __has_builtin(__builtin_amdgcn_wmma_f32_16x16x4_f32)
#define TTT_HAVE_WMMA_F32 1
#endif
#endif
#ifndef TTT_HAVE_WMMA_F32
#define TTT_HAVE_WMMA_F32 0
#endif

#if TTT_HAVE_WMMA_F32
typedef v2f FragA4;
typedef v2f FragB4;
// A(m,k): p = &mat[m*ld + k0], k stride 1.  element i <-> K = i + 2*half
__device__ inline FragA4 loadA4(const float* p, int half, float sgn) {
  v2f a; a.x = sgn * p[2*half]; a.y = sgn * p[2*half + 1]; return a;
}
// B(k,n): p = &mat[k0*ld + n].  element i <-> K = i + 2*half
__device__ inline FragB4 loadB4(const float* p, int ld, int half) {
  v2f b; b.x = p[(2*half)*ld]; b.y = p[(2*half + 1)*ld]; return b;
}
__device__ inline v8f wmma4(FragA4 a, FragB4 b, v8f c) {
  return __builtin_amdgcn_wmma_f32_16x16x4_f32(false, a, false, b, (short)0, c,
                                               false, false);
}
#else
typedef v16bf FragA4;
typedef v16bf FragB4;
// Zero-padded bf16 emulation of a K=4 f32 WMMA step.
__device__ inline FragA4 loadA4(const float* p, int half, float sgn) {
  v16bf v;
  for (int e = 0; e < 16; e++) v[e] = (__bf16)0.0f;
  if (half == 0) for (int e = 0; e < 4; e++) v[e] = (__bf16)(sgn * p[e]);
  return v;
}
__device__ inline FragB4 loadB4(const float* p, int ld, int half) {
  v16bf v;
  for (int e = 0; e < 16; e++) v[e] = (__bf16)0.0f;
  if (half == 0) for (int e = 0; e < 4; e++) v[e] = (__bf16)(p[e*ld]);
  return v;
}
__device__ inline v8f wmma4(FragA4 a, FragB4 b, v8f c) {
  return wmma_bf16x32(a, b, c);
}
#endif

// ------- async global->LDS path (guarded; fallback = register staging) ----
#if defined(__has_builtin)
#if __has_builtin(__builtin_amdgcn_global_load_async_to_lds_b128) && \
    __has_builtin(__builtin_amdgcn_s_wait_asynccnt)
#define TTT_HAVE_ASYNC_LDS 1
#endif
#endif
#ifndef TTT_HAVE_ASYNC_LDS
#define TTT_HAVE_ASYNC_LDS 0
#endif

#if TTT_HAVE_ASYNC_LDS
// Builtin expects v4i ("int __vector_size__(16)") pointers: global source in
// addrspace(1), LDS destination in addrspace(3).
typedef __attribute__((__vector_size__(4 * sizeof(int)))) int v4i;
typedef __attribute__((address_space(1))) v4i as1_v4i;
typedef __attribute__((address_space(3))) v4i as3_v4i;
__device__ inline as1_v4i* to_glb(const void* p) {
  return (as1_v4i*)(unsigned long long)p;
}
// LDS aperture lives in the high 32 bits of a generic pointer; low 32 bits
// are the in-allocation LDS byte offset (flat-addressing rules).
__device__ inline as3_v4i* to_lds(const void* p) {
  return (as3_v4i*)(unsigned int)(unsigned long long)p;
}
#endif

// ---------------------------------------------------------------- kernels

__global__ void cvt_f32_bf16_kernel(const float* __restrict__ in,
                                    u16* __restrict__ out, int n) {
  int i = blockIdx.x * blockDim.x + threadIdx.x;
  if (i < n) out[i] = f32_to_bf16(in[i]);
}

// C[M,N] = A[M,K] @ W[N,K]^T     (bf16 in, f32 out)
union FragU { v16bf v; uint4 q[2]; };

#if TTT_HAVE_ASYNC_LDS
// Double-buffered: GLOBAL_LOAD_ASYNC_TO_LDS_B128 copies the next K-slab while
// WMMAs consume the current one; ASYNCcnt drains before the buffer swap.
__global__ __launch_bounds__(256)
void gemm_bf16_kernel(const u16* __restrict__ A, const u16* __restrict__ W,
                      float* __restrict__ C, int M, int N, int Kd) {
  constexpr int BM = 128, BN = 128, BK = 64, LDK = BK + 8;
  __shared__ u16 sA[2][BM * LDK];
  __shared__ u16 sB[2][BN * LDK];
  const int tid  = (int)threadIdx.x;
  const int lane = tid & 31;
  const int wvid = tid >> 5;
  const int half = lane >> 4;
  const int n16  = lane & 15;
  const int wm   = wvid >> 1;         // 0..3 (M)
  const int wn   = wvid & 1;          // 0..1 (N)
  const int row0 = (int)blockIdx.y * BM;
  const int col0 = (int)blockIdx.x * BN;

  v8f zero_ = {};
  v8f acc[2][4];
  for (int i = 0; i < 2; i++)
    for (int t = 0; t < 4; t++) acc[i][t] = zero_;

  const int ldrow = tid >> 1;          // 0..127
  const int ldcol = (tid & 1) * 32;    // 0 or 32
  const u16* gA = A + (size_t)(row0 + ldrow) * Kd + ldcol;
  const u16* gB = W + (size_t)(col0 + ldrow) * Kd + ldcol;

  // issue one K-slab copy (each thread moves 32+32 elements as 4+4 b128s)
  auto issue_stage = [&](int k0, int buf) {
    const u16* ga = gA + k0;
    const u16* gb = gB + k0;
    u16* la = &sA[buf][ldrow * LDK + ldcol];
    u16* lb = &sB[buf][ldrow * LDK + ldcol];
    for (int j = 0; j < 4; j++) {
      __builtin_amdgcn_global_load_async_to_lds_b128(to_glb(ga + 8 * j),
                                                     to_lds(la + 8 * j), 0, 0);
      __builtin_amdgcn_global_load_async_to_lds_b128(to_glb(gb + 8 * j),
                                                     to_lds(lb + 8 * j), 0, 0);
    }
  };

  issue_stage(0, 0);
  __builtin_amdgcn_s_wait_asynccnt(0);
  __syncthreads();

  int buf = 0;
  for (int k0 = 0; k0 < Kd; k0 += BK) {
    if (k0 + BK < Kd) issue_stage(k0 + BK, buf ^ 1);   // overlap with compute

    for (int kk = 0; kk < BK; kk += 32) {
      FragU bfr[4];
      for (int t = 0; t < 4; t++) {
        const u16* p = &sB[buf][(wn * 64 + t * 16 + n16) * LDK + kk + 16 * half];
        bfr[t].q[0] = ((const uint4*)p)[0];
        bfr[t].q[1] = ((const uint4*)p)[1];
      }
      FragU afr[2];
      for (int i = 0; i < 2; i++) {
        const u16* p = &sA[buf][(wm * 32 + i * 16 + n16) * LDK + kk];
        afr[i].q[0] = *(const uint4*)(p + 16 * half);
        afr[i].q[1] = *(const uint4*)(p + 16 * half + 8);
      }
      for (int i = 0; i < 2; i++)
        for (int t = 0; t < 4; t++)
          acc[i][t] = wmma_bf16x32(afr[i].v, bfr[t].v, acc[i][t]);
    }

    __builtin_amdgcn_s_wait_asynccnt(0);  // next slab resident
    __syncthreads();                      // everyone done reading this slab
    buf ^= 1;
  }

  for (int i = 0; i < 2; i++)
    for (int t = 0; t < 4; t++)
      for (int j = 0; j < 8; j++) {
        int r = row0 + wm * 32 + i * 16 + j + 8 * half;
        int c = col0 + wn * 64 + t * 16 + n16;
        C[(size_t)r * N + c] = acc[i][t][j];
      }
}
#else
__global__ __launch_bounds__(256)
void gemm_bf16_kernel(const u16* __restrict__ A, const u16* __restrict__ W,
                      float* __restrict__ C, int M, int N, int Kd) {
  constexpr int BM = 128, BN = 128, BK = 64, LDK = BK + 8;
  __shared__ u16 sA[BM * LDK];
  __shared__ u16 sB[BN * LDK];
  const int tid  = (int)threadIdx.x;
  const int lane = tid & 31;
  const int wvid = tid >> 5;
  const int half = lane >> 4;
  const int n16  = lane & 15;
  const int wm   = wvid >> 1;
  const int wn   = wvid & 1;
  const int row0 = (int)blockIdx.y * BM;
  const int col0 = (int)blockIdx.x * BN;

  v8f zero_ = {};
  v8f acc[2][4];
  for (int i = 0; i < 2; i++)
    for (int t = 0; t < 4; t++) acc[i][t] = zero_;

  const int ldrow = tid >> 1;
  const int ldcol = (tid & 1) * 32;
  const u16* gA = A + (size_t)(row0 + ldrow) * Kd + ldcol;
  const u16* gB = W + (size_t)(col0 + ldrow) * Kd + ldcol;

  for (int k0 = 0; k0 < Kd; k0 += BK) {
    uint4 ra0 = ((const uint4*)(gA + k0))[0];
    uint4 ra1 = ((const uint4*)(gA + k0))[1];
    uint4 ra2 = ((const uint4*)(gA + k0))[2];
    uint4 ra3 = ((const uint4*)(gA + k0))[3];
    uint4 rb0 = ((const uint4*)(gB + k0))[0];
    uint4 rb1 = ((const uint4*)(gB + k0))[1];
    uint4 rb2 = ((const uint4*)(gB + k0))[2];
    uint4 rb3 = ((const uint4*)(gB + k0))[3];
    if (k0 + BK < Kd) {                    // WGP-scope prefetch of next slab
      __builtin_prefetch(gA + k0 + BK, 0, 3);
      __builtin_prefetch(gB + k0 + BK, 0, 3);
    }
    __syncthreads();
    uint4* dA = (uint4*)(sA + ldrow * LDK + ldcol);
    dA[0] = ra0; dA[1] = ra1; dA[2] = ra2; dA[3] = ra3;
    uint4* dB = (uint4*)(sB + ldrow * LDK + ldcol);
    dB[0] = rb0; dB[1] = rb1; dB[2] = rb2; dB[3] = rb3;
    __syncthreads();

    for (int kk = 0; kk < BK; kk += 32) {
      FragU bfr[4];
      for (int t = 0; t < 4; t++) {
        const u16* p = sB + (wn * 64 + t * 16 + n16) * LDK + kk + 16 * half;
        bfr[t].q[0] = ((const uint4*)p)[0];
        bfr[t].q[1] = ((const uint4*)p)[1];
      }
      FragU afr[2];
      for (int i = 0; i < 2; i++) {
        const u16* p = sA + (wm * 32 + i * 16 + n16) * LDK + kk;
        afr[i].q[0] = *(const uint4*)(p + 16 * half);
        afr[i].q[1] = *(const uint4*)(p + 16 * half + 8);
      }
      for (int i = 0; i < 2; i++)
        for (int t = 0; t < 4; t++)
          acc[i][t] = wmma_bf16x32(afr[i].v, bfr[t].v, acc[i][t]);
    }
  }

  for (int i = 0; i < 2; i++)
    for (int t = 0; t < 4; t++)
      for (int j = 0; j < 8; j++) {
        int r = row0 + wm * 32 + i * 16 + j + 8 * half;
        int c = col0 + wn * 64 + t * 16 + n16;
        C[(size_t)r * N + c] = acc[i][t][j];
      }
}
#endif

// Interleaved RoPE applied in-place to XQ and XK (net effect of
// permute_qk -> rotary(half-split) -> undo_permute_qk).
__global__ void rope_kernel(float* __restrict__ XQ, float* __restrict__ XK) {
  int idx = blockIdx.x * blockDim.x + threadIdx.x;       // over B*L*D/2 pairs
  if (idx >= cB * cL * (cD / 2)) return;
  int pair = idx & (cD / 2 - 1);       // 0..1023
  int row  = idx / (cD / 2);           // b*L + l
  int l    = row & (cL - 1);
  int h    = pair >> 5;                // head
  int i    = pair & 31;                // pair index within head
  float inv = __expf(-((2.0f * (float)i) / (float)cHD) * 9.210340371976184f);
  float ang = (float)l * inv;
  float s, c;
  __sincosf(ang, &s, &c);
  size_t base = (size_t)row * cD + h * cHD + 2 * i;
  float q0 = XQ[base], q1 = XQ[base + 1];
  XQ[base]     = q0 * c - q1 * s;
  XQ[base + 1] = q1 * c + q0 * s;
  float k0 = XK[base], k1 = XK[base + 1];
  XK[base]     = k0 * c - k1 * s;
  XK[base + 1] = k1 * c + k0 * s;
}

// lr[(b*NH+h)*L + l] = sigmoid(hidden[b,l,:] . lrw[h,:] + lrb[h]) / HD
__global__ __launch_bounds__(256)
void lr_kernel(const float* __restrict__ hidden, const float* __restrict__ lrw,
               const float* __restrict__ lrb, float* __restrict__ lr) {
  __shared__ float srow[cD];
  int row = blockIdx.x;                 // b*L + l
  for (int c = threadIdx.x; c < cD; c += 256)
    srow[c] = hidden[(size_t)row * cD + c];
  __syncthreads();
  int wvid = threadIdx.x >> 5, lane = threadIdx.x & 31;
  int b = row / cL, l = row % cL;
  for (int h = wvid; h < cNH; h += 8) {
    float s = 0.f;
    for (int d = lane; d < cD; d += 32) s += srow[d] * lrw[(size_t)h * cD + d];
    for (int m = 16; m; m >>= 1) s += __shfl_xor(s, m, 32);
    if (lane == 0) {
      float v = s + lrb[h];
      float sg = 1.0f / (1.0f + __expf(-v));
      lr[((size_t)b * cNH + h) * cL + l] = sg / (float)cHD;
    }
  }
}

// One wave per (b,h): sequential TTT scan over 128 chunks.
__global__ __launch_bounds__(32)
void ttt_scan_kernel(float* __restrict__ XQ,            // in xq / out y
                     const float* __restrict__ XK, const float* __restrict__ XV,
                     const float* __restrict__ lrbuf,   // (B*NH)*L
                     const float* __restrict__ W1g,     // NH*HD*HD
                     const float* __restrict__ b1g,     // NH*HD
                     const float* __restrict__ ltok,    // K
                     const float* __restrict__ gwg,     // NH*HD
                     const float* __restrict__ gbg) {   // NH*HD
  __shared__ float sW1[cHD * cHD];
  __shared__ float sB1[cHD];
  __shared__ float sXQ[cK * cHD];
  __shared__ float sXK[cK * cHD];
  __shared__ float sXV[cK * cHD];
  __shared__ float sGr[cK * cHD];
  __shared__ float sAco[cK * cK];
  __shared__ float sAupd[cHD * cK];
  __shared__ float sLr[cK];
  __shared__ float sTok[cK];

  const int bh = blockIdx.x;
  const int b = bh / cNH, h = bh % cNH;
  const int lane = (int)threadIdx.x;
  const int half = lane >> 4;
  const int n16  = lane & 15;

  for (int i = lane; i < cHD * cHD; i += 32) sW1[i] = W1g[(size_t)h * cHD * cHD + i];
  for (int i = lane; i < cHD; i += 32) sB1[i] = b1g[h * cHD + i];
  if (lane < cK) {
    float t = 1.0f / (float)(lane + 1) + ltok[lane];
    sTok[lane] = t > 0.f ? t : 0.f;
  }
  float gam[4], bet[4];
  for (int t = 0; t < 4; t++) {
    gam[t] = gwg[h * cHD + t * 16 + n16];
    bet[t] = gbg[h * cHD + t * 16 + n16];
  }
  __syncthreads();

  for (int nc = 0; nc < cNC; nc++) {
    const int l0 = nc * cK;
    for (int i = lane; i < cK * cHD; i += 32) {
      int r = i >> 6, c = i & 63;
      size_t g = ((size_t)(b * cL + l0 + r)) * cD + h * cHD + c;
      sXQ[i] = XQ[g];
      sXK[i] = XK[g];
      sXV[i] = XV[g];
    }
    if (lane < cK) sLr[lane] = lrbuf[(size_t)bh * cL + l0 + lane];
    __syncthreads();
    const float lrv = sLr[n16];

    // Z1 = xk@W1 + b1 ; Zq = xq@W1
    v8f accK[4], accQ[4];
    for (int t = 0; t < 4; t++) {
      v8f ck = {}; v8f cq = {};
      for (int kq = 0; kq < cHD; kq += 4) {
        FragB4 fb = loadB4(&sW1[kq * cHD + t * 16 + n16], cHD, half);
        FragA4 fk = loadA4(&sXK[n16 * cHD + kq], half, 1.0f);
        FragA4 fq = loadA4(&sXQ[n16 * cHD + kq], half, 1.0f);
        ck = wmma4(fk, fb, ck);
        cq = wmma4(fq, fb, cq);
      }
      float bv = sB1[t * 16 + n16];
      for (int j = 0; j < 8; j++) ck[j] += bv;
      accK[t] = ck; accQ[t] = cq;
    }

    // ln_fused_l2_bwd row stats (over 64 features)
    float mu[8], rst[8];
    for (int j = 0; j < 8; j++) {
      float s  = accK[0][j] + accK[1][j] + accK[2][j] + accK[3][j];
      float s2 = accK[0][j]*accK[0][j] + accK[1][j]*accK[1][j]
               + accK[2][j]*accK[2][j] + accK[3][j]*accK[3][j];
      s = redsum16(s); s2 = redsum16(s2);
      float m_ = s * (1.0f / cHD);
      float v_ = s2 * (1.0f / cHD) - m_ * m_;
      mu[j] = m_;
      rst[j] = rsqrtf(v_ + 1e-6f);
    }
    float gs[8], gxs[8];
    for (int j = 0; j < 8; j++) { gs[j] = 0.f; gxs[j] = 0.f; }
    for (int t = 0; t < 4; t++)
      for (int j = 0; j < 8; j++) {
        int r = j + 8 * half;
        float xh = (accK[t][j] - mu[j]) * rst[j];
        float tv = sXV[r * cHD + t * 16 + n16] - sXK[r * cHD + t * 16 + n16];
        float g  = (gam[t] * xh + bet[t] - tv) * gam[t];
        gs[j] += g; gxs[j] += g * xh;
      }
    for (int j = 0; j < 8; j++) {
      gs[j]  = redsum16(gs[j])  * (1.0f / cHD);
      gxs[j] = redsum16(gxs[j]) * (1.0f / cHD);
    }
    for (int t = 0; t < 4; t++)
      for (int j = 0; j < 8; j++) {
        int r = j + 8 * half;
        float xh = (accK[t][j] - mu[j]) * rst[j];
        float tv = sXV[r * cHD + t * 16 + n16] - sXK[r * cHD + t * 16 + n16];
        float g  = (gam[t] * xh + bet[t] - tv) * gam[t];
        sGr[r * cHD + t * 16 + n16] = (g - gs[j] - xh * gxs[j]) * rst[j];
      }
    __syncthreads();

    // Attn = xq @ xk^T  (raw, pre-tril)
    v8f attn = {};
    for (int kq = 0; kq < cHD; kq += 4) {
      FragA4 fa = loadA4(&sXQ[n16 * cHD + kq], half, 1.0f);
      FragB4 fb = loadB4(&sXK[n16 * cHD + kq], 1, half);
      attn = wmma4(fa, fb, attn);
    }
    // Acoef = tril( et .* (Attn_raw + 1) ),  et[i,j] = tok[i]*lr[j]
    for (int j = 0; j < 8; j++) {
      int r = j + 8 * half;
      float v = 0.f;
      if (r >= n16) v = sTok[r] * lrv * (attn[j] + 1.0f);
      sAco[r * cK + n16] = v;
    }
    __syncthreads();

    // Z1_bar = Zq - Acoef@grad + b1 ; out = xq + ln_fwd(Z1_bar)
    for (int t = 0; t < 4; t++) {
      v8f cq = accQ[t];
      for (int kq = 0; kq < cK; kq += 4) {
        FragA4 fa = loadA4(&sAco[n16 * cK + kq], half, -1.0f);
        FragB4 fb = loadB4(&sGr[kq * cHD + t * 16 + n16], cHD, half);
        cq = wmma4(fa, fb, cq);
      }
      float bv = sB1[t * 16 + n16];
      for (int j = 0; j < 8; j++) cq[j] += bv;
      accQ[t] = cq;
    }
    for (int j = 0; j < 8; j++) {
      float s  = accQ[0][j] + accQ[1][j] + accQ[2][j] + accQ[3][j];
      float s2 = accQ[0][j]*accQ[0][j] + accQ[1][j]*accQ[1][j]
               + accQ[2][j]*accQ[2][j] + accQ[3][j]*accQ[3][j];
      s = redsum16(s); s2 = redsum16(s2);
      float m_ = s * (1.0f / cHD);
      float v_ = s2 * (1.0f / cHD) - m_ * m_;
      mu[j] = m_;
      rst[j] = rsqrtf(v_ + 1e-6f);
    }
    for (int t = 0; t < 4; t++)
      for (int j = 0; j < 8; j++) {
        int r = j + 8 * half;
        float xh = (accQ[t][j] - mu[j]) * rst[j];
        float o  = sXQ[r * cHD + t * 16 + n16] + gam[t] * xh + bet[t];
        size_t g = ((size_t)(b * cL + l0 + r)) * cD + h * cHD + t * 16 + n16;
        XQ[g] = o;
      }

    // ---- state update: W1 -= (last_eta*xk)^T @ grad ; b1 -= sum(last_eta*grad)
    const float tok15 = sTok[cK - 1];
    for (int i = lane; i < cHD * cK; i += 32) {
      int d = i >> 4, j = i & 15;
      sAupd[i] = -(tok15 * sLr[j]) * sXK[j * cHD + d];
    }
    for (int e = lane; e < cHD; e += 32) {
      float s = 0.f;
      for (int j = 0; j < cK; j++) s += sLr[j] * sGr[j * cHD + e];
      sB1[e] -= tok15 * s;
    }
    __syncthreads();
    for (int mt = 0; mt < 4; mt++)
      for (int t = 0; t < 4; t++) {
        v8f cw = {};
        for (int j = 0; j < 8; j++)
          cw[j] = sW1[(mt * 16 + j + 8 * half) * cHD + t * 16 + n16];
        for (int kq = 0; kq < cK; kq += 4) {
          FragA4 fa = loadA4(&sAupd[(mt * 16 + n16) * cK + kq], half, 1.0f);
          FragB4 fb = loadB4(&sGr[kq * cHD + t * 16 + n16], cHD, half);
          cw = wmma4(fa, fb, cw);
        }
        for (int j = 0; j < 8; j++)
          sW1[(mt * 16 + j + 8 * half) * cHD + t * 16 + n16] = cw[j];
      }
    __syncthreads();
  }
}

// Post-norm over D=2048, write bf16 for the final projection.
__global__ __launch_bounds__(256)
void postnorm_kernel(const float* __restrict__ Y, const float* __restrict__ pw,
                     const float* __restrict__ pb, u16* __restrict__ out) {
  __shared__ float sS[8], sS2[8];
  int row = blockIdx.x;
  const float* yr = Y + (size_t)row * cD;
  float s = 0.f, s2 = 0.f;
  for (int c = threadIdx.x; c < cD; c += 256) {
    float v = yr[c];
    s += v; s2 += v * v;
  }
  for (int m = 16; m; m >>= 1) { s += __shfl_xor(s, m, 32); s2 += __shfl_xor(s2, m, 32); }
  if ((threadIdx.x & 31) == 0) { sS[threadIdx.x >> 5] = s; sS2[threadIdx.x >> 5] = s2; }
  __syncthreads();
  s = 0.f; s2 = 0.f;
  for (int i = 0; i < 8; i++) { s += sS[i]; s2 += sS2[i]; }
  float mu = s / (float)cD;
  float var = s2 / (float)cD - mu * mu;
  float rstd = rsqrtf(var + 1e-5f);
  for (int c = threadIdx.x; c < cD; c += 256) {
    float v = (yr[c] - mu) * rstd * pw[c] + pb[c];
    out[(size_t)row * cD + c] = f32_to_bf16(v);
  }
}

// ---------------------------------------------------------------- launcher

extern "C" void kernel_launch(void* const* d_in, const int* in_sizes, int n_in,
                              void* d_out, int out_size, void* d_ws, size_t ws_size,
                              hipStream_t stream) {
  (void)in_sizes; (void)n_in; (void)out_size; (void)ws_size;
  const float* hidden = (const float*)d_in[0];
  const float* wq  = (const float*)d_in[1];
  const float* wk  = (const float*)d_in[2];
  const float* wv  = (const float*)d_in[3];
  const float* wo  = (const float*)d_in[4];
  const float* lrw = (const float*)d_in[5];
  const float* lrb = (const float*)d_in[6];
  const float* ltok = (const float*)d_in[7];
  const float* W1g = (const float*)d_in[8];
  const float* b1g = (const float*)d_in[9];
  const float* tnw = (const float*)d_in[10];
  const float* tnb = (const float*)d_in[11];
  const float* pnw = (const float*)d_in[12];
  const float* pnb = (const float*)d_in[13];
  float* out = (float*)d_out;

  char* p = (char*)d_ws;
  u16* hid_bf = (u16*)p;  p += (size_t)cB * cL * cD * 2;   // reused for post-norm bf16
  u16* wq_bf  = (u16*)p;  p += (size_t)cD * cD * 2;
  u16* wk_bf  = (u16*)p;  p += (size_t)cD * cD * 2;
  u16* wv_bf  = (u16*)p;  p += (size_t)cD * cD * 2;
  u16* wo_bf  = (u16*)p;  p += (size_t)cD * cD * 2;
  float* XQ   = (float*)p; p += (size_t)cB * cL * cD * 4;  // also scan output Y
  float* XK   = (float*)p; p += (size_t)cB * cL * cD * 4;
  float* XV   = (float*)p; p += (size_t)cB * cL * cD * 4;
  float* lrbuf = (float*)p; p += (size_t)cB * cNH * cL * 4;

  const int M = cB * cL;

  // 1) f32 -> bf16 conversions
  int nh = cB * cL * cD;
  cvt_f32_bf16_kernel<<<(nh + 255) / 256, 256, 0, stream>>>(hidden, hid_bf, nh);
  int nw = cD * cD;
  cvt_f32_bf16_kernel<<<(nw + 255) / 256, 256, 0, stream>>>(wq, wq_bf, nw);
  cvt_f32_bf16_kernel<<<(nw + 255) / 256, 256, 0, stream>>>(wk, wk_bf, nw);
  cvt_f32_bf16_kernel<<<(nw + 255) / 256, 256, 0, stream>>>(wv, wv_bf, nw);
  cvt_f32_bf16_kernel<<<(nw + 255) / 256, 256, 0, stream>>>(wo, wo_bf, nw);

  // 2) projections (A @ W^T)
  dim3 gg(cD / 128, M / 128), gb(256);
  gemm_bf16_kernel<<<gg, gb, 0, stream>>>(hid_bf, wq_bf, XQ, M, cD, cD);
  gemm_bf16_kernel<<<gg, gb, 0, stream>>>(hid_bf, wk_bf, XK, M, cD, cD);
  gemm_bf16_kernel<<<gg, gb, 0, stream>>>(hid_bf, wv_bf, XV, M, cD, cD);

  // 3) RoPE on XQ, XK
  int npairs = cB * cL * cD / 2;
  rope_kernel<<<(npairs + 255) / 256, 256, 0, stream>>>(XQ, XK);

  // 4) per-token learning rates
  lr_kernel<<<cB * cL, 256, 0, stream>>>(hidden, lrw, lrb, lrbuf);

  // 5) TTT scan (one wave32 per (b,h)), writes output in-place over XQ
  ttt_scan_kernel<<<cB * cNH, 32, 0, stream>>>(XQ, XK, XV, lrbuf, W1g, b1g,
                                               ltok, tnw, tnb);

  // 6) post-norm -> bf16 (reuses hid_bf)
  postnorm_kernel<<<cB * cL, 256, 0, stream>>>(XQ, pnw, pnb, hid_bf);

  // 7) output projection -> d_out
  gemm_bf16_kernel<<<gg, gb, 0, stream>>>(hid_bf, wo_bf, out, M, cD, cD);
}
